// SparseAttention_34359738368598
// MI455X (gfx1250) — compile-verified, matrix operand order checked by reading, not run on hardware
//
#include <hip/hip_runtime.h>
#include <hip/hip_bf16.h>

// ---------------------------------------------------------------------------
// Types for WMMA operands
// ---------------------------------------------------------------------------
typedef __attribute__((ext_vector_type(16))) __bf16         v16bf;
typedef __attribute__((ext_vector_type(8)))  float          v8f;
typedef __attribute__((ext_vector_type(8)))  unsigned short v8u;
typedef __attribute__((ext_vector_type(16))) unsigned short v16u;
typedef __attribute__((ext_vector_type(4)))  int            v4i;

// ---------------------------------------------------------------------------
// CDNA5 async global->LDS copy (ASYNCcnt-tracked), with sync fallback
// ---------------------------------------------------------------------------
#if defined(__has_builtin)
#if __has_builtin(__builtin_amdgcn_global_load_async_to_lds_b128)
#define HAS_ASYNC_LDS 1
#endif
#endif
#ifndef HAS_ASYNC_LDS
#define HAS_ASYNC_LDS 0
#endif

__device__ __forceinline__ void async_cp16(const unsigned short* g,
                                           unsigned short* l) {
#if HAS_ASYNC_LDS
    __builtin_amdgcn_global_load_async_to_lds_b128(
        (__attribute__((address_space(1))) v4i*)g,
        (__attribute__((address_space(3))) v4i*)l, 0, 0);
#else
    *(v8u*)l = *(const v8u*)g;
#endif
}

__device__ __forceinline__ void async_wait0() {
#if HAS_ASYNC_LDS
#if __has_builtin(__builtin_amdgcn_s_wait_asynccnt)
    __builtin_amdgcn_s_wait_asynccnt(0);
#else
    asm volatile("s_wait_asynccnt 0x0" ::: "memory");
#endif
#endif
}

// float -> bf16 bit pattern (round-to-nearest-even), avoids __bf16 arithmetic
__device__ __forceinline__ unsigned short f2bf(float f) {
    union { float f; unsigned u; } x; x.f = f;
    unsigned u = x.u;
    unsigned r = (u + 0x7fffu + ((u >> 16) & 1u)) >> 16;
    return (unsigned short)r;
}

// Load a 16x32 bf16 A/B fragment from row-major LDS (per ISA 7.12.2):
// lane L: row = row0 + (L & 15); 16 elems = k0+kb+0..7 and k0+kb+16..23,
// kb = (L < 16) ? 0 : 8.  Both chunks are 16-byte aligned (ld multiple of 8).
__device__ __forceinline__ v16bf frag_rm(const unsigned short* p, int ld,
                                         int row0, int k0) {
    int lane = threadIdx.x & 31;
    int r  = row0 + (lane & 15);
    int kb = k0 + ((lane & 16) >> 1);           // +8 for upper half-wave
    const unsigned short* base = p + r * ld + kb;
    v8u lo = *(const v8u*)(base);
    v8u hi = *(const v8u*)(base + 16);
    v16u t;
#pragma unroll
    for (int i = 0; i < 8; ++i) { t[i] = lo[i]; t[i + 8] = hi[i]; }
    return __builtin_bit_cast(v16bf, t);
}

// ---------------------------------------------------------------------------
// Elementwise f32 -> bf16 conversion (one-shot prep so GEMM staging can use
// pure byte-copy async DMA).  4 elements / thread.
// ---------------------------------------------------------------------------
__global__ __launch_bounds__(256) void cvt_f32_bf16(
    const float* __restrict__ s, unsigned short* __restrict__ d, int n) {
    int i = (blockIdx.x * 256 + threadIdx.x) * 4;
    if (i >= n) return;
    float4 v = *(const float4*)(s + i);
    ushort4 o;
    o.x = f2bf(v.x); o.y = f2bf(v.y); o.z = f2bf(v.z); o.w = f2bf(v.w);
    *(ushort4*)(d + i) = o;
}

// ---------------------------------------------------------------------------
// Async double-buffered GEMM:  C[M x 1024] = A[M x 1024] * W[1024 x 1024]^T
// A and W are bf16 in memory; HBM -> LDS via GLOBAL_LOAD_ASYNC_TO_LDS_B128
// (ping-pong buffers, DMA of tile k+32 overlaps WMMA on tile k).
// 64x64 workgroup tile, 8 waves x two 16x16 output tiles.
// ---------------------------------------------------------------------------
template <bool OUT_BF16>
__global__ __launch_bounds__(256) void gemm_bf16_async(
    const unsigned short* __restrict__ A, const unsigned short* __restrict__ W,
    void* __restrict__ Cptr, int K) {
    __shared__ unsigned short At[2][64 * 40];
    __shared__ unsigned short Wt[2][64 * 40];

    const int m0   = blockIdx.x * 64;
    const int n0   = blockIdx.y * 64;
    const int t    = threadIdx.x;
    const int wave = t >> 5;
    const int lane = t & 31;

    v8f acc[2] = {};

    const int row = t >> 2;        // 0..63
    const int c0  = (t & 3) * 8;   // 0,8,16,24

    // prologue: stage first k-slab
    async_cp16(A + (size_t)(m0 + row) * K + c0, &At[0][row * 40 + c0]);
    async_cp16(W + (size_t)(n0 + row) * K + c0, &Wt[0][row * 40 + c0]);
    async_wait0();
    __syncthreads();

    int cur = 0;
    for (int kt = 0; kt < K; kt += 32) {
        // start DMA of the next slab into the other buffer
        if (kt + 32 < K) {
            async_cp16(A + (size_t)(m0 + row) * K + kt + 32 + c0,
                       &At[cur ^ 1][row * 40 + c0]);
            async_cp16(W + (size_t)(n0 + row) * K + kt + 32 + c0,
                       &Wt[cur ^ 1][row * 40 + c0]);
        }
        // compute on the current buffer
#pragma unroll
        for (int s = 0; s < 2; ++s) {
            int tile = wave * 2 + s;
            int mi = tile >> 2, ni = tile & 3;
            v16bf a = frag_rm(At[cur], 40, mi * 16, 0);
            v16bf b = frag_rm(Wt[cur], 40, ni * 16, 0);
            acc[s] = __builtin_amdgcn_wmma_f32_16x16x32_bf16(
                false, a, false, b, (short)0, acc[s], false, false);
        }
        async_wait0();
        __syncthreads();
        cur ^= 1;
    }

    // store C fragments: col = lane&15, rows = r + 8*(lane>=16)
#pragma unroll
    for (int s = 0; s < 2; ++s) {
        int tile = wave * 2 + s;
        int mi = tile >> 2, ni = tile & 3;
        int col = n0 + ni * 16 + (lane & 15);
        int rb  = m0 + mi * 16 + ((lane & 16) >> 1);
#pragma unroll
        for (int r = 0; r < 8; ++r) {
            int rowg = rb + r;
            if (OUT_BF16)
                ((unsigned short*)Cptr)[(size_t)rowg * 1024 + col] = f2bf(acc[s][r]);
            else
                ((float*)Cptr)[(size_t)rowg * 1024 + col] = acc[s][r];
        }
    }
}

// ---------------------------------------------------------------------------
// Fallback GEMM (round-1 path): in-kernel f32->bf16 staging; used when the
// workspace is too small for pre-converted bf16 copies of X and the weights.
// ---------------------------------------------------------------------------
template <bool A_BF16, bool OUT_BF16>
__global__ __launch_bounds__(256) void gemm_bf16_wmma(
    const void* __restrict__ Aptr, const float* __restrict__ W,
    void* __restrict__ Cptr, int K) {
    __shared__ unsigned short At[64 * 40];
    __shared__ unsigned short Wt[64 * 40];

    const int m0   = blockIdx.x * 64;
    const int n0   = blockIdx.y * 64;
    const int t    = threadIdx.x;
    const int wave = t >> 5;
    const int lane = t & 31;

    v8f acc[2] = {};
    const int row = t >> 2;
    const int c0  = (t & 3) * 8;

    for (int kt = 0; kt < K; kt += 32) {
        if (A_BF16) {
            const unsigned short* A = (const unsigned short*)Aptr;
            *(v8u*)(At + row * 40 + c0) =
                *(const v8u*)(A + (size_t)(m0 + row) * K + kt + c0);
        } else {
            const float* src = (const float*)Aptr + (size_t)(m0 + row) * K + kt + c0;
            v8u av;
#pragma unroll
            for (int i = 0; i < 8; ++i) av[i] = f2bf(src[i]);
            *(v8u*)(At + row * 40 + c0) = av;
        }
        {
            const float* src = W + (size_t)(n0 + row) * K + kt + c0;
            v8u wv;
#pragma unroll
            for (int i = 0; i < 8; ++i) wv[i] = f2bf(src[i]);
            *(v8u*)(Wt + row * 40 + c0) = wv;
        }
        __syncthreads();
#pragma unroll
        for (int s = 0; s < 2; ++s) {
            int tile = wave * 2 + s;
            int mi = tile >> 2, ni = tile & 3;
            v16bf a = frag_rm(At, 40, mi * 16, 0);
            v16bf b = frag_rm(Wt, 40, ni * 16, 0);
            acc[s] = __builtin_amdgcn_wmma_f32_16x16x32_bf16(
                false, a, false, b, (short)0, acc[s], false, false);
        }
        __syncthreads();
    }
#pragma unroll
    for (int s = 0; s < 2; ++s) {
        int tile = wave * 2 + s;
        int mi = tile >> 2, ni = tile & 3;
        int col = n0 + ni * 16 + (lane & 15);
        int rb  = m0 + mi * 16 + ((lane & 16) >> 1);
#pragma unroll
        for (int r = 0; r < 8; ++r) {
            int rowg = rb + r;
            if (OUT_BF16)
                ((unsigned short*)Cptr)[(size_t)rowg * 1024 + col] = f2bf(acc[s][r]);
            else
                ((float*)Cptr)[(size_t)rowg * 1024 + col] = acc[s][r];
        }
    }
}

// ---------------------------------------------------------------------------
// Block-sparse attention with ALiBi, flash-style streaming softmax.
// One workgroup per (b, h, qb): 64 queries x (11 blocks of 64 keys).
// Q/K tiles staged via async DMA; V staged with an on-the-fly transpose.
// ---------------------------------------------------------------------------
__global__ __launch_bounds__(256) void sparse_attn_wmma(
    const unsigned short* __restrict__ Qb,
    const unsigned short* __restrict__ Kb,
    const unsigned short* __restrict__ Vb,
    const float* __restrict__ slopes,
    const int* __restrict__ bidx,
    unsigned short* __restrict__ Ob) {
    __shared__ unsigned short Qs[64 * 72];
    __shared__ unsigned short Ks[64 * 72];
    __shared__ unsigned short Vt[64 * 72];   // V^T: [d][j]
    __shared__ unsigned short Ps[64 * 72];   // softmaxed probs (bf16)
    __shared__ float          Ss[64 * 68];   // raw scores (f32)
    __shared__ float rowM[64], rowL[64], rowScale[64];

    const int qb = blockIdx.x;   // query block 0..63
    const int h  = blockIdx.y;   // head 0..15
    const int b  = blockIdx.z;   // batch 0..1
    const int t  = threadIdx.x;
    const int wave = t >> 5, lane = t & 31;
    const float slope = slopes[h];

    const int rowt = t >> 2, ct0 = (t & 3) * 16;

    // Load Q tile (64 x 64 bf16) for this (b, qb, h) via async DMA
    {
        const unsigned short* src =
            Qb + ((size_t)(b * 4096 + qb * 64 + rowt)) * 1024 + h * 64 + ct0;
        async_cp16(src,     Qs + rowt * 72 + ct0);
        async_cp16(src + 8, Qs + rowt * 72 + ct0 + 8);
    }
    if (t < 64) { rowM[t] = -3.0e38f; rowL[t] = 0.0f; }
    v8f oacc[2] = {};
    const int tile0 = wave * 2;
    async_wait0();
    __syncthreads();

    for (int mblk = 0; mblk < 11; ++mblk) {
        int kidx = bidx[qb * 11 + mblk];
        if (kidx < 0) continue;   // masked block: exp(-1e9) ~ 0, skip

        // Gather K (async, row-major) and V (transposed) blocks into LDS
        {
            const size_t gro =
                ((size_t)(b * 4096 + kidx * 64 + rowt)) * 1024 + h * 64 + ct0;
            async_cp16(Kb + gro,     Ks + rowt * 72 + ct0);
            async_cp16(Kb + gro + 8, Ks + rowt * 72 + ct0 + 8);
            const unsigned short* vsrc = Vb + gro;
            v8u v0 = *(const v8u*)(vsrc);
            v8u v1 = *(const v8u*)(vsrc + 8);
#pragma unroll
            for (int i = 0; i < 8; ++i) {
                Vt[(ct0 + i) * 72 + rowt]     = v0[i];
                Vt[(ct0 + 8 + i) * 72 + rowt] = v1[i];
            }
        }
        async_wait0();
        __syncthreads();

        // S = (Q K^T) * SCALE + alibi bias, via WMMA (K-dim 64 = 2 steps)
#pragma unroll
        for (int s = 0; s < 2; ++s) {
            int tile = tile0 + s;
            int mi = tile >> 2, ni = tile & 3;
            v8f sacc = {};
#pragma unroll
            for (int kk = 0; kk < 64; kk += 32) {
                v16bf a  = frag_rm(Qs, 72, mi * 16, kk);
                v16bf bb = frag_rm(Ks, 72, ni * 16, kk);
                sacc = __builtin_amdgcn_wmma_f32_16x16x32_bf16(
                    false, a, false, bb, (short)0, sacc, false, false);
            }
            int col = ni * 16 + (lane & 15);
            int rb  = mi * 16 + ((lane & 16) >> 1);
            int kpos = kidx * 64 + col;
#pragma unroll
            for (int r = 0; r < 8; ++r) {
                int rowq = rb + r;
                int qpos = qb * 64 + rowq;
                float sv = sacc[r] * 0.125f - slope * fabsf((float)(qpos - kpos));
                Ss[rowq * 68 + col] = sv;
            }
        }
        __syncthreads();

        // Online softmax row update (one thread per query row)
        if (t < 64) {
            float m = rowM[t];
#pragma unroll 4
            for (int c = 0; c < 64; ++c) m = fmaxf(m, Ss[t * 68 + c]);
            float sc = __expf(rowM[t] - m);
            float l  = rowL[t] * sc;
#pragma unroll 4
            for (int c = 0; c < 64; ++c) {
                float p = __expf(Ss[t * 68 + c] - m);
                l += p;
                Ps[t * 72 + c] = f2bf(p);
            }
            rowM[t] = m; rowL[t] = l; rowScale[t] = sc;
        }
        __syncthreads();

        // Rescale O and accumulate O += P * V via WMMA
#pragma unroll
        for (int s = 0; s < 2; ++s) {
            int tile = tile0 + s;
            int mi = tile >> 2, ni = tile & 3;
            int rb = mi * 16 + ((lane & 16) >> 1);
#pragma unroll
            for (int r = 0; r < 8; ++r) oacc[s][r] *= rowScale[rb + r];
#pragma unroll
            for (int kk = 0; kk < 64; kk += 32) {
                v16bf a  = frag_rm(Ps, 72, mi * 16, kk);
                v16bf bb = frag_rm(Vt, 72, ni * 16, kk);   // B[j][d] = Vt[d][j]
                oacc[s] = __builtin_amdgcn_wmma_f32_16x16x32_bf16(
                    false, a, false, bb, (short)0, oacc[s], false, false);
            }
        }
        __syncthreads();
    }

    // Normalize and write attn output (bf16, layout [b, l, h*64 + d])
#pragma unroll
    for (int s = 0; s < 2; ++s) {
        int tile = tile0 + s;
        int mi = tile >> 2, ni = tile & 3;
        int col = ni * 16 + (lane & 15);
        int rb  = mi * 16 + ((lane & 16) >> 1);
#pragma unroll
        for (int r = 0; r < 8; ++r) {
            int rowq = rb + r;
            float invl = 1.0f / rowL[rowq];
            Ob[((size_t)(b * 4096 + qb * 64 + rowq)) * 1024 + h * 64 + col] =
                f2bf(oacc[s][r] * invl);
        }
    }
}

// ---------------------------------------------------------------------------
// Launch: convert to bf16 -> QKV projections -> sparse attention -> out proj
// ---------------------------------------------------------------------------
extern "C" void kernel_launch(void* const* d_in, const int* in_sizes, int n_in,
                              void* d_out, int out_size, void* d_ws, size_t ws_size,
                              hipStream_t stream) {
    (void)in_sizes; (void)n_in; (void)out_size;

    const float* X      = (const float*)d_in[0];
    const float* Wq     = (const float*)d_in[1];
    const float* Wk     = (const float*)d_in[2];
    const float* Wv     = (const float*)d_in[3];
    const float* Wo     = (const float*)d_in[4];
    const float* slopes = (const float*)d_in[5];
    const int*   bidx   = (const int*)d_in[6];

    unsigned char* ws = (unsigned char*)d_ws;
    const size_t SEG  = (size_t)16 * 1024 * 1024;  // 8192*1024 bf16
    const size_t WSEG = (size_t)2 * 1024 * 1024;   // 1024*1024 bf16

    unsigned short* Qb   = (unsigned short*)(ws);
    unsigned short* Kb   = (unsigned short*)(ws + 1 * SEG);
    unsigned short* Vb   = (unsigned short*)(ws + 2 * SEG);
    unsigned short* XbOb = (unsigned short*)(ws + 3 * SEG);  // Xb, later Ob

    dim3 gg(128, 16);            // M/64 = 8192/64, N/64 = 1024/64
    dim3 ga(64, 16, 2);          // (qb, h, b)

    if (ws_size >= 4 * SEG + 4 * WSEG) {
        // bf16 workspace path: pre-convert, then async-DMA GEMMs
        unsigned short* Wqb = (unsigned short*)(ws + 4 * SEG + 0 * WSEG);
        unsigned short* Wkb = (unsigned short*)(ws + 4 * SEG + 1 * WSEG);
        unsigned short* Wvb = (unsigned short*)(ws + 4 * SEG + 2 * WSEG);
        unsigned short* Wob = (unsigned short*)(ws + 4 * SEG + 3 * WSEG);

        cvt_f32_bf16<<<8192, 256, 0, stream>>>(X,  XbOb, 8192 * 1024);
        cvt_f32_bf16<<<1024, 256, 0, stream>>>(Wq, Wqb, 1024 * 1024);
        cvt_f32_bf16<<<1024, 256, 0, stream>>>(Wk, Wkb, 1024 * 1024);
        cvt_f32_bf16<<<1024, 256, 0, stream>>>(Wv, Wvb, 1024 * 1024);
        cvt_f32_bf16<<<1024, 256, 0, stream>>>(Wo, Wob, 1024 * 1024);

        gemm_bf16_async<true><<<gg, 256, 0, stream>>>(XbOb, Wqb, Qb, 1024);
        gemm_bf16_async<true><<<gg, 256, 0, stream>>>(XbOb, Wkb, Kb, 1024);
        gemm_bf16_async<true><<<gg, 256, 0, stream>>>(XbOb, Wvb, Vb, 1024);

        // Xb is dead now; slot 3 becomes Ob
        sparse_attn_wmma<<<ga, 256, 0, stream>>>(Qb, Kb, Vb, slopes, bidx, XbOb);

        gemm_bf16_async<false><<<gg, 256, 0, stream>>>(XbOb, Wob,
                                                       (float*)d_out, 1024);
    } else {
        // minimal-workspace fallback: convert inside the GEMM
        gemm_bf16_wmma<false, true><<<gg, 256, 0, stream>>>(X, Wq, Qb, 1024);
        gemm_bf16_wmma<false, true><<<gg, 256, 0, stream>>>(X, Wk, Kb, 1024);
        gemm_bf16_wmma<false, true><<<gg, 256, 0, stream>>>(X, Wv, Vb, 1024);

        sparse_attn_wmma<<<ga, 256, 0, stream>>>(XbOb ? Qb : Qb, Kb, Vb, slopes, bidx, XbOb);

        gemm_bf16_wmma<true, false><<<gg, 256, 0, stream>>>(XbOb, Wo,
                                                            (float*)d_out, 1024);
    }
}
